// NodeGraphMatchingModule_17059610100117
// MI455X (gfx1250) — matrix operand order repaired and checked by reading, not compiled, not run on hardware
//
#include <hip/hip_runtime.h>
#include <hip/hip_bf16.h>

// Problem sizes (from reference): L=4096, D=512, P=64, H=128.
constexpr int Lc = 4096;
constexpr int Dc = 512;
constexpr int Pc = 64;
constexpr int Hc = 128;
#define EPSF 1e-8f
#define AB_STRIDE 40  // padded LDS row stride in halves (16B-aligned, bank-friendly)

typedef __attribute__((ext_vector_type(16))) __bf16 v16bf;
typedef __attribute__((ext_vector_type(8)))  float  v8f;

union Frag16 {
  v16bf v;
  unsigned short s[16];
  uint4 q[2];
};

__device__ __forceinline__ unsigned short f2bf(float x) {
  unsigned int u = __float_as_uint(x);
  u += 0x7FFFu + ((u >> 16) & 1u);  // round-to-nearest-even
  return (unsigned short)(u >> 16);
}

__device__ __forceinline__ v8f wmma_bf16(const Frag16& a, const Frag16& b, v8f c) {
  return __builtin_amdgcn_wmma_f32_16x16x32_bf16(false, a.v, false, b.v,
                                                 (short)0, c, false, false);
}

// CDNA5 async global->LDS copy (16B per lane), tracked with ASYNCcnt.
// lds = 32-bit LDS byte address (low 32 bits of the generic shared pointer),
// g   = per-lane global address.
__device__ __forceinline__ void async_copy_b128(unsigned lds, const void* g) {
  asm volatile("global_load_async_to_lds_b128 %0, %1, off"
               :: "v"(lds), "v"(g) : "memory");
}

// ---------------------------------------------------------------- utilities
__global__ void zero_kernel(float* p, int n) {
  int i = blockIdx.x * blockDim.x + threadIdx.x;
  if (i < n) p[i] = 0.f;
}

// fp32 features -> bf16 copies + row L2 norms. grid = 2*L blocks, 256 thr.
__global__ __launch_bounds__(256) void prep_kernel(
    const float* __restrict__ fp, const float* __restrict__ fh,
    unsigned short* __restrict__ pb, unsigned short* __restrict__ hb,
    float* __restrict__ np, float* __restrict__ nh) {
  __shared__ float red[256];
  int b = blockIdx.x;
  int row = b & (Lc - 1);
  const float* src = (b < Lc) ? fp : fh;
  unsigned short* dstb = (b < Lc) ? pb : hb;
  float* dstn = (b < Lc) ? np : nh;
  int t = threadIdx.x;
  float x0 = src[(size_t)row * Dc + t];
  float x1 = src[(size_t)row * Dc + t + 256];
  dstb[(size_t)row * Dc + t] = f2bf(x0);
  dstb[(size_t)row * Dc + t + 256] = f2bf(x1);
  red[t] = x0 * x0 + x1 * x1;
  __syncthreads();
  for (int s = 128; s > 0; s >>= 1) {
    if (t < s) red[t] += red[t + s];
    __syncthreads();
  }
  if (t == 0) dstn[row] = sqrtf(red[0]);
}

// att = (P @ H^T) / clamp(|p||h|), stored bf16; also fp32 row/col sums of att.
// 128x128 tile per block; 8 waves, each 32x64 (2x4 accumulators).
// Double-buffered LDS staging via async global->LDS copies (ASYNCcnt pipeline):
// DMA of tile k+1 overlaps the 8 WMMAs of tile k.
__global__ __launch_bounds__(256) void attn_kernel(
    const unsigned short* __restrict__ Pb, const unsigned short* __restrict__ Hb,
    const float* __restrict__ np, const float* __restrict__ nh,
    unsigned short* __restrict__ att, float* __restrict__ rowsum,
    float* __restrict__ colsum) {
  __shared__ unsigned short lA[2][128 * AB_STRIDE];
  __shared__ unsigned short lB[2][128 * AB_STRIDE];
  const int bi = blockIdx.y * 128;
  const int bj = blockIdx.x * 128;
  const int t = threadIdx.x;
  const int lane = t & 31;
  const int wave = t >> 5;
  const int wr = (wave & 3) * 32;
  const int wc = (wave >> 2) * 64;
  const int ln = lane & 15;
  const int lk8  = (lane >> 4) << 3;   // A fragment k-offset (0 / 8)
  const int lk16 = (lane >> 4) << 4;   // unused; B fragment also uses lk8 (0 / 8)
  const int lr  = t >> 1;
  const int lch = (t & 1) << 4;

  const unsigned short* gA = Pb + (size_t)(bi + lr) * Dc + lch;
  const unsigned short* gB = Hb + (size_t)(bj + lr) * Dc + lch;

  auto issue = [&](int bsel, int k0) {
    unsigned la = (unsigned)(size_t)&lA[bsel][lr * AB_STRIDE + lch];
    unsigned lb = (unsigned)(size_t)&lB[bsel][lr * AB_STRIDE + lch];
    async_copy_b128(la,      gA + k0);      // 8 halves
    async_copy_b128(la + 16, gA + k0 + 8);  // next 8 halves
    async_copy_b128(lb,      gB + k0);
    async_copy_b128(lb + 16, gB + k0 + 8);
  };

  v8f acc[2][4] = {};
  int buf = 0;
  issue(0, 0);
  for (int k0 = 0; k0 < Dc; k0 += 32) {
    if (k0 + 32 < Dc) {
      issue(buf ^ 1, k0 + 32);
      // 4 newest (next-buffer) ops may stay outstanding; async loads complete
      // in order, so <=4 means the current buffer's 4 ops have landed.
      asm volatile("s_wait_asynccnt 4" ::: "memory");
    } else {
      asm volatile("s_wait_asynccnt 0" ::: "memory");
    }
    __syncthreads();

    Frag16 fa[2], fb[4];
#pragma unroll
    for (int r = 0; r < 2; ++r) {
      int off = (wr + r * 16 + ln) * AB_STRIDE + lk8;
      fa[r].q[0] = *reinterpret_cast<const uint4*>(&lA[buf][off]);
      fa[r].q[1] = *reinterpret_cast<const uint4*>(&lA[buf][off + 16]);
    }
#pragma unroll
    for (int c = 0; c < 4; ++c) {
      int off = (wc + c * 16 + ln) * AB_STRIDE + lk8;
      fb[c].q[0] = *reinterpret_cast<const uint4*>(&lB[buf][off]);
      fb[c].q[1] = *reinterpret_cast<const uint4*>(&lB[buf][off + 16]);
    }
#pragma unroll
    for (int r = 0; r < 2; ++r)
#pragma unroll
      for (int c = 0; c < 4; ++c)
        acc[r][c] = wmma_bf16(fa[r], fb[c], acc[r][c]);

    __syncthreads();  // all waves done reading buf before it is re-filled
    buf ^= 1;
  }

  // Epilogue: normalize, store bf16, accumulate row/col sums.
  const int hi8 = (lane >> 4) << 3;
  float nhv[4];
  int colg[4];
#pragma unroll
  for (int c = 0; c < 4; ++c) {
    colg[c] = bj + wc + c * 16 + ln;
    nhv[c] = nh[colg[c]];
  }
  float colAcc[4] = {0.f, 0.f, 0.f, 0.f};
#pragma unroll
  for (int r = 0; r < 2; ++r) {
#pragma unroll
    for (int i = 0; i < 8; ++i) {
      int row = bi + wr + r * 16 + hi8 + i;
      float npv = np[row];
      float rsum = 0.f;
#pragma unroll
      for (int c = 0; c < 4; ++c) {
        float d = npv * nhv[c];
        float a = acc[r][c][i] / (d > EPSF ? d : EPSF);
        att[(size_t)row * Lc + colg[c]] = f2bf(a);
        rsum += a;
        colAcc[c] += a;
      }
      rsum += __shfl_xor(rsum, 1, 32);
      rsum += __shfl_xor(rsum, 2, 32);
      rsum += __shfl_xor(rsum, 4, 32);
      rsum += __shfl_xor(rsum, 8, 32);
      if (ln == 0) atomicAdd(&rowsum[row], rsum);
    }
  }
#pragma unroll
  for (int c = 0; c < 4; ++c) atomicAdd(&colsum[colg[c]], colAcc[c]);
}

// out[row,d] = (A @ B)[row,d] / clamp(denom[row]).  A = att (bf16, LxL),
// optionally transposed during LDS staging; B = feature (bf16, LxD) staged
// transposed so LDS_B[n=d][k=j] = F[j, d].
__global__ __launch_bounds__(256) void attmean_kernel(
    const unsigned short* __restrict__ A, const unsigned short* __restrict__ B,
    const float* __restrict__ denom, float* __restrict__ out, int transA) {
  __shared__ unsigned short lA[128 * AB_STRIDE];
  __shared__ unsigned short lB[128 * AB_STRIDE];
  const int bi = blockIdx.y * 128;
  const int d0 = blockIdx.x * 128;
  const int t = threadIdx.x;
  const int lane = t & 31;
  const int wave = t >> 5;
  const int wr = (wave & 3) * 32;
  const int wc = (wave >> 2) * 64;
  const int ln = lane & 15;
  const int lk8  = (lane >> 4) << 3;
  const int lk16 = (lane >> 4) << 4;
  const int lr  = t >> 1;
  const int lch = (t & 1) << 4;

  v8f acc[2][4] = {};

  for (int k0 = 0; k0 < Lc; k0 += 32) {
    __syncthreads();
    if (!transA) {
      const uint4* ga = reinterpret_cast<const uint4*>(A + (size_t)(bi + lr) * Lc + k0 + lch);
      uint4 a0 = ga[0], a1 = ga[1];
      *reinterpret_cast<uint4*>(&lA[lr * AB_STRIDE + lch])     = a0;
      *reinterpret_cast<uint4*>(&lA[lr * AB_STRIDE + lch + 8]) = a1;
    } else {
      int k = t >> 3, ms = (t & 7) << 4;
      const uint4* ga = reinterpret_cast<const uint4*>(A + (size_t)(k0 + k) * Lc + bi + ms);
      uint4 a0 = ga[0], a1 = ga[1];
      unsigned short tmp[16];
      *reinterpret_cast<uint4*>(tmp)     = a0;
      *reinterpret_cast<uint4*>(tmp + 8) = a1;
#pragma unroll
      for (int e = 0; e < 16; ++e) lA[(ms + e) * AB_STRIDE + k] = tmp[e];
    }
    {
      int k = t >> 3, ns = (t & 7) << 4;
      const uint4* gb = reinterpret_cast<const uint4*>(B + (size_t)(k0 + k) * Dc + d0 + ns);
      uint4 b0 = gb[0], b1 = gb[1];
      unsigned short tmp[16];
      *reinterpret_cast<uint4*>(tmp)     = b0;
      *reinterpret_cast<uint4*>(tmp + 8) = b1;
#pragma unroll
      for (int e = 0; e < 16; ++e) lB[(ns + e) * AB_STRIDE + k] = tmp[e];
    }
    __syncthreads();

    Frag16 fa[2], fb[4];
#pragma unroll
    for (int r = 0; r < 2; ++r) {
      int off = (wr + r * 16 + ln) * AB_STRIDE + lk8;
      fa[r].q[0] = *reinterpret_cast<const uint4*>(&lA[off]);
      fa[r].q[1] = *reinterpret_cast<const uint4*>(&lA[off + 16]);
    }
#pragma unroll
    for (int c = 0; c < 4; ++c) {
      int off = (wc + c * 16 + ln) * AB_STRIDE + lk8;
      fb[c].q[0] = *reinterpret_cast<const uint4*>(&lB[off]);
      fb[c].q[1] = *reinterpret_cast<const uint4*>(&lB[off + 16]);
    }
#pragma unroll
    for (int r = 0; r < 2; ++r)
#pragma unroll
      for (int c = 0; c < 4; ++c)
        acc[r][c] = wmma_bf16(fa[r], fb[c], acc[r][c]);
  }

  const int hi8 = (lane >> 4) << 3;
#pragma unroll
  for (int r = 0; r < 2; ++r)
#pragma unroll
    for (int i = 0; i < 8; ++i) {
      int row = bi + wr + r * 16 + hi8 + i;
      float d = denom[row];
      float inv = 1.f / (d > EPSF ? d : EPSF);
#pragma unroll
      for (int c = 0; c < 4; ++c) {
        int col = d0 + wc + c * 16 + ln;
        out[(size_t)row * Dc + col] = acc[r][c][i] * inv;
      }
    }
}

// Multi-perspective weighted cosine: out[l,p] = num / (max(n1,eps)*max(n2,eps))
// num = (v1*v2)@w2^T, n1=sqrt((v1*v1)@w2^T), n2=sqrt((v2*v2)@w2^T), w2=mp_w^2.
// One block = 128 rows; wave = 16 rows x all 64 P columns; 12 WMMA accums.
__global__ __launch_bounds__(256) void match_kernel(
    const float* __restrict__ v1, const float* __restrict__ v2,
    const float* __restrict__ mpw, float* __restrict__ out) {
  const int t = threadIdx.x;
  const int lane = t & 31;
  const int wave = t >> 5;
  const int ln = lane & 15;
  const int r0 = blockIdx.x * 128 + wave * 16;
  const int m = r0 + ln;               // A-matrix row for this lane
  const int lk8  = (lane >> 4) << 3;
  const int lk16 = (lane >> 4) << 4;

  v8f accN[4] = {}, acc1[4] = {}, acc2[4] = {};

  for (int k0 = 0; k0 < Dc; k0 += 32) {
    const float4* p1 = reinterpret_cast<const float4*>(v1 + (size_t)m * Dc + k0 + lk8);
    const float4* p2 = reinterpret_cast<const float4*>(v2 + (size_t)m * Dc + k0 + lk8);
    float4 x[4], y[4];
    x[0] = p1[0]; x[1] = p1[1]; x[2] = p1[4]; x[3] = p1[5];
    y[0] = p2[0]; y[1] = p2[1]; y[2] = p2[4]; y[3] = p2[5];
    Frag16 fN, f1, f2;
#pragma unroll
    for (int q = 0; q < 4; ++q) {
      float xs[4] = {x[q].x, x[q].y, x[q].z, x[q].w};
      float ys[4] = {y[q].x, y[q].y, y[q].z, y[q].w};
#pragma unroll
      for (int e = 0; e < 4; ++e) {
        int idx = q * 4 + e;
        fN.s[idx] = f2bf(xs[e] * ys[e]);
        f1.s[idx] = f2bf(xs[e] * xs[e]);
        f2.s[idx] = f2bf(ys[e] * ys[e]);
      }
    }
#pragma unroll
    for (int c = 0; c < 4; ++c) {
      int p = c * 16 + ln;
      const float4* pw = reinterpret_cast<const float4*>(mpw + (size_t)p * Dc + k0 + lk8);
      float4 w0 = pw[0], w1 = pw[1], w2v = pw[4], w3 = pw[5];
      float ws[16] = {w0.x, w0.y, w0.z, w0.w, w1.x, w1.y, w1.z, w1.w,
                      w2v.x, w2v.y, w2v.z, w2v.w, w3.x, w3.y, w3.z, w3.w};
      Frag16 fb;
#pragma unroll
      for (int e = 0; e < 16; ++e) fb.s[e] = f2bf(ws[e] * ws[e]);
      accN[c] = wmma_bf16(fN, fb, accN[c]);
      acc1[c] = wmma_bf16(f1, fb, acc1[c]);
      acc2[c] = wmma_bf16(f2, fb, acc2[c]);
    }
  }
  const int hi8 = (lane >> 4) << 3;
#pragma unroll
  for (int c = 0; c < 4; ++c) {
    int col = c * 16 + ln;
#pragma unroll
    for (int i = 0; i < 8; ++i) {
      int row = r0 + hi8 + i;
      float n1 = sqrtf(fmaxf(acc1[c][i], 0.f));
      float n2 = sqrtf(fmaxf(acc2[c][i], 0.f));
      float den = fmaxf(n1, EPSF) * fmaxf(n2, EPSF);
      out[(size_t)row * Pc + col] = accN[c][i] / den;
    }
  }
}

// gx[dir][t][j] = dot(x_dir[t], w_ih[j,:]) + b_ih[j] + b_hh[j]
// dir: 0 = (match_p, fwd), 1 = (match_p, rev), 2 = (match_h, fwd), 3 = (match_h, rev)
__global__ __launch_bounds__(512) void gx_kernel(
    const float* __restrict__ match_p, const float* __restrict__ match_h,
    const float* __restrict__ wih_f, const float* __restrict__ wih_r,
    const float* __restrict__ bih_f, const float* __restrict__ bhh_f,
    const float* __restrict__ bih_r, const float* __restrict__ bhh_r,
    float* __restrict__ gx) {
  __shared__ float xs[Pc];
  const int tstep = blockIdx.x;
  const int dir = blockIdx.y;
  const int j = threadIdx.x;
  const float* src = (dir < 2) ? match_p : match_h;
  const int trow = (dir & 1) ? (Lc - 1 - tstep) : tstep;
  if (j < Pc) xs[j] = src[(size_t)trow * Pc + j];
  __syncthreads();
  const float* wih = (dir & 1) ? wih_r : wih_f;
  const float* bi = (dir & 1) ? bih_r : bih_f;
  const float* bh = (dir & 1) ? bhh_r : bhh_f;
  float z = bi[j] + bh[j];
  const float* wrow = wih + (size_t)j * Pc;
#pragma unroll
  for (int k = 0; k < Pc; ++k) z += xs[k] * wrow[k];
  gx[((size_t)dir * Lc + tstep) * (4 * Hc) + j] = z;
}

// Transpose w_hh [4H,H] -> [H,4H] for coalesced LSTM matvec. grid = 2.
__global__ __launch_bounds__(256) void whhT_kernel(
    const float* __restrict__ whh_f, const float* __restrict__ whh_r,
    float* __restrict__ wT) {
  const float* src = blockIdx.x ? whh_r : whh_f;
  float* dst = wT + (size_t)blockIdx.x * (4 * Hc * Hc);
  for (int idx = threadIdx.x; idx < 4 * Hc * Hc; idx += 256) {
    int jj = idx >> 7;   // 0..511
    int kk = idx & 127;  // 0..127
    dst[(size_t)kk * (4 * Hc) + jj] = src[(size_t)jj * Hc + kk];
  }
}

// Sequential LSTM, one block per direction. h,c,z in LDS; 4096 steps.
__global__ __launch_bounds__(512) void lstm_kernel(
    const float* __restrict__ gx, const float* __restrict__ wT,
    float* __restrict__ outp) {
  __shared__ float hsh[Hc], csh[Hc], zsh[4 * Hc];
  const int dir = blockIdx.x;
  const int j = threadIdx.x;
  if (j < Hc) { hsh[j] = 0.f; csh[j] = 0.f; }
  __syncthreads();
  const float* g = gx + (size_t)dir * Lc * (4 * Hc);
  const float* w = wT + ((dir & 1) ? (size_t)(4 * Hc * Hc) : (size_t)0);
  for (int tt = 0; tt < Lc; ++tt) {
    float z = g[(size_t)tt * (4 * Hc) + j];
#pragma unroll 8
    for (int k = 0; k < Hc; ++k) z += hsh[k] * w[(size_t)k * (4 * Hc) + j];
    zsh[j] = z;
    __syncthreads();
    if (j < Hc) {
      float zi = zsh[j], zf = zsh[Hc + j], zg = zsh[2 * Hc + j], zo = zsh[3 * Hc + j];
      float si = 1.f / (1.f + expf(-zi));
      float sf = 1.f / (1.f + expf(-zf));
      float so = 1.f / (1.f + expf(-zo));
      float cc = sf * csh[j] + si * tanhf(zg);
      csh[j] = cc;
      hsh[j] = so * tanhf(cc);
    }
    __syncthreads();
  }
  if (j < Hc) outp[dir * Hc + j] = hsh[j];
}

extern "C" void kernel_launch(void* const* d_in, const int* in_sizes, int n_in,
                              void* d_out, int out_size, void* d_ws, size_t ws_size,
                              hipStream_t stream) {
  (void)in_sizes; (void)n_in; (void)out_size; (void)ws_size;
  const float* feature_p = (const float*)d_in[0];
  const float* feature_h = (const float*)d_in[1];
  const float* mp_w   = (const float*)d_in[2];
  const float* w_ih_f = (const float*)d_in[3];
  const float* w_hh_f = (const float*)d_in[4];
  const float* b_ih_f = (const float*)d_in[5];
  const float* b_hh_f = (const float*)d_in[6];
  const float* w_ih_r = (const float*)d_in[7];
  const float* w_hh_r = (const float*)d_in[8];
  const float* b_ih_r = (const float*)d_in[9];
  const float* b_hh_r = (const float*)d_in[10];
  float* out = (float*)d_out;

  char* ws = (char*)d_ws;
  size_t off = 0;
  auto alloc = [&](size_t bytes) -> void* {
    void* p = ws + off;
    off += (bytes + 255) & ~(size_t)255;
    return p;
  };
  unsigned short* Pb = (unsigned short*)alloc((size_t)Lc * Dc * 2);
  unsigned short* Hb = (unsigned short*)alloc((size_t)Lc * Dc * 2);
  float* np      = (float*)alloc((size_t)Lc * 4);
  float* nh      = (float*)alloc((size_t)Lc * 4);
  float* rowsum  = (float*)alloc((size_t)Lc * 4);
  float* colsum  = (float*)alloc((size_t)Lc * 4);
  unsigned short* att = (unsigned short*)alloc((size_t)Lc * Lc * 2);
  float* amh     = (float*)alloc((size_t)Lc * Dc * 4);
  float* amp     = (float*)alloc((size_t)Lc * Dc * 4);
  float* match_p = (float*)alloc((size_t)Lc * Pc * 4);
  float* match_h = (float*)alloc((size_t)Lc * Pc * 4);
  float* gx      = (float*)alloc((size_t)4 * Lc * 4 * Hc * 4);
  float* wT      = (float*)alloc((size_t)2 * 4 * Hc * Hc * 4);

  // rowsum/colsum are contiguous (each Lc*4 bytes, 256B-aligned): zero both.
  zero_kernel<<<(2 * Lc + 255) / 256, 256, 0, stream>>>(rowsum, 2 * Lc);
  prep_kernel<<<2 * Lc, 256, 0, stream>>>(feature_p, feature_h, Pb, Hb, np, nh);

  dim3 g1(Lc / 128, Lc / 128);
  attn_kernel<<<g1, 256, 0, stream>>>(Pb, Hb, np, nh, att, rowsum, colsum);

  dim3 g2(Dc / 128, Lc / 128);
  attmean_kernel<<<g2, 256, 0, stream>>>(att, Hb, rowsum, amh, 0);
  attmean_kernel<<<g2, 256, 0, stream>>>(att, Pb, colsum, amp, 1);

  match_kernel<<<Lc / 128, 256, 0, stream>>>(feature_p, amh, mp_w, match_p);
  match_kernel<<<Lc / 128, 256, 0, stream>>>(feature_h, amp, mp_w, match_h);

  dim3 g3(Lc, 4);
  gx_kernel<<<g3, 512, 0, stream>>>(match_p, match_h, w_ih_f, w_ih_r,
                                    b_ih_f, b_hh_f, b_ih_r, b_hh_r, gx);
  whhT_kernel<<<2, 256, 0, stream>>>(w_hh_f, w_hh_r, wT);
  lstm_kernel<<<4, 512, 0, stream>>>(gx, wT, out);
}
